// BasicResidualSEBlock_30030411334438
// MI455X (gfx1250) — matrix operand order repaired and checked by reading, not compiled, not run on hardware
//
#include <hip/hip_runtime.h>
#include <hip/hip_bf16.h>
#include <math.h>

// ---------------------------------------------------------------------------
// Shapes
// ---------------------------------------------------------------------------
#define BN   8
#define CN   64
#define HN   256
#define WN   256
#define NJ   129          // min(W/2,H/2)+1 crop count
#define EPSV 1e-5f

typedef __attribute__((ext_vector_type(16))) _Float16 v16h;
typedef __attribute__((ext_vector_type(8)))  _Float16 v8h;
typedef __attribute__((ext_vector_type(8)))  float    v8f;

#define SHUF16(lo, hi) __builtin_shufflevector((lo), (hi), 0,1,2,3,4,5,6,7,8,9,10,11,12,13,14,15)

// ---------------------------------------------------------------------------
// Workspace layout (bytes). Total ~203 MB.
// ---------------------------------------------------------------------------
static constexpr size_t align256(size_t x) { return (x + 255) & ~size_t(255); }
static constexpr size_t SZ_ACT   = (size_t)BN * HN * WN * CN * 2;   // NHWC f16 activations
static constexpr size_t SZ_WP    = (size_t)9 * CN * CN * 2;         // packed weights f16
static constexpr size_t SZ_RP    = (size_t)BN * HN * CN * 4;        // rowpart / colpart f32
static constexpr size_t SZ_MEANS = (size_t)BN * NJ * CN * 4;

static constexpr size_t OFF_XH    = 0;
static constexpr size_t OFF_R1    = OFF_XH  + SZ_ACT;
static constexpr size_t OFF_R2    = OFF_R1  + SZ_ACT;
static constexpr size_t OFF_WP1   = OFF_R2  + SZ_ACT;
static constexpr size_t OFF_WP2   = OFF_WP1 + align256(SZ_WP);
static constexpr size_t OFF_B1    = OFF_WP2 + align256(SZ_WP);
static constexpr size_t OFF_B2    = OFF_B1  + 256;
static constexpr size_t OFF_ROWP  = OFF_B2  + 256;
static constexpr size_t OFF_COLP  = OFF_ROWP + SZ_RP;
static constexpr size_t OFF_MEANS = OFF_COLP + SZ_RP;
static constexpr size_t OFF_GATE  = OFF_MEANS + align256(SZ_MEANS);

// ---------------------------------------------------------------------------
// 1a. Pack weights: OIHW f32 -> [tap][c_out][c_in] f16 with BN scale folded.
// ---------------------------------------------------------------------------
__global__ __launch_bounds__(256) void pack_w_kernel(
    const float* __restrict__ w, const float* __restrict__ bn_g,
    const float* __restrict__ bn_v, _Float16* __restrict__ wp) {
  int idx = blockIdx.x * 256 + threadIdx.x;           // 9*64*64 = 36864
  if (idx >= 9 * CN * CN) return;
  int ci  = idx & 63;
  int co  = (idx >> 6) & 63;
  int tap = idx >> 12;
  float scale = bn_g[co] * rsqrtf(bn_v[co] + EPSV);
  wp[((size_t)tap * CN + co) * CN + ci] =
      (_Float16)(w[((size_t)co * CN + ci) * 9 + tap] * scale);
}

// 1b. Fold conv bias + BN shift into one per-channel bias.
__global__ __launch_bounds__(64) void fold_bias_kernel(
    const float* __restrict__ cb, const float* __restrict__ g,
    const float* __restrict__ bb, const float* __restrict__ m,
    const float* __restrict__ v, float* __restrict__ bias) {
  int co = threadIdx.x;
  float s = g[co] * rsqrtf(v[co] + EPSV);
  bias[co] = cb[co] * s + bb[co] - m[co] * s;
}

// ---------------------------------------------------------------------------
// 2. NCHW f32 -> NHWC f16 (LDS transpose, coalesced both directions).
//    One block per (b, h).
// ---------------------------------------------------------------------------
__global__ __launch_bounds__(256) void nchw_to_nhwc_kernel(
    const float* __restrict__ x, _Float16* __restrict__ xh) {
  __shared__ _Float16 lds[CN * (WN + 2)];             // pad row to 258 halves
  int b = blockIdx.x >> 8;
  int h = blockIdx.x & 255;
  int tid = threadIdx.x;
  const float* src = x + (((size_t)b * CN) * HN + h) * WN;
#pragma unroll 4
  for (int it = 0; it < 64; ++it) {
    int idx = it * 256 + tid;
    int c = idx >> 8;                                  // coalesced along w
    int w = idx & 255;
    lds[c * (WN + 2) + w] = (_Float16)src[(size_t)c * HN * WN + w];
  }
  __syncthreads();
  _Float16* dst = xh + (((size_t)b * HN + h) * WN) * CN;
#pragma unroll 4
  for (int it = 0; it < 64; ++it) {
    int idx = it * 256 + tid;
    int w = idx >> 6;                                  // coalesced along c
    int c = idx & 63;
    dst[idx] = lds[c * (WN + 2) + w];
  }
}

// ---------------------------------------------------------------------------
// 3. 3x3 conv as implicit GEMM via v_wmma_f32_16x16x32_f16.
//    GEMM: M = pixels (one output row of 128 px per block half), N = 64 c_out,
//    K = 576 (9 taps x 64 c_in), folded scale in weights, bias+ReLU epilogue.
//    Block: 256 threads = 8 waves; wave w owns pixel rows [16w, 16w+16).
// ---------------------------------------------------------------------------
__global__ __launch_bounds__(256) void conv3x3_wmma_kernel(
    const _Float16* __restrict__ in,    // [B][H][W][C] f16
    const _Float16* __restrict__ wp,    // [9][co][ci] f16 (B^T layout)
    const float*    __restrict__ bias,  // [64]
    _Float16*       __restrict__ out) { // [B][H][W][C] f16
  __shared__ _Float16 Alds[128 * 32];   // 8 KB staging: 128 px x 32 K-halves

  int blk  = blockIdx.x;
  int wblk = blk & 1;
  int h    = (blk >> 1) & 255;
  int b    = blk >> 9;
  int tid  = threadIdx.x;
  int lane = tid & 31;
  int wave = tid >> 5;
  int w0   = wblk * 128;

  int n0 = lane & 15;        // N (c_out) within 16-block
  int kq = lane >> 4;        // which K-half of the fragment this lane owns

  v8f acc[4] = {v8f{}, v8f{}, v8f{}, v8f{}};

  for (int tap = 0; tap < 9; ++tap) {
    int dh = tap / 3 - 1;
    int dw = tap % 3 - 1;
    int hs = h + dh;
    bool hok = (hs >= 0) & (hs < HN);
#pragma unroll
    for (int half = 0; half < 2; ++half) {
      // ---- stage A tile: 128 pixels x 32 c_in halves (zero-pad halo) ----
      int p    = tid >> 1;
      int ws_  = w0 + p + dw;
      int coff = half * 32 + (tid & 1) * 16;
      v8h z0 = v8h{}, z1 = v8h{};
      if (hok && ws_ >= 0 && ws_ < WN) {
        const _Float16* src =
            in + (((size_t)b * HN + hs) * WN + ws_) * CN + coff;
        z0 = ((const v8h*)src)[0];
        z1 = ((const v8h*)src)[1];
        __builtin_prefetch(src + (size_t)WN * CN, 0, 1);  // next-row prefetch
      }
      __syncthreads();
      *(v8h*)&Alds[p * 32 + (tid & 1) * 16 + 0] = z0;
      *(v8h*)&Alds[p * 32 + (tid & 1) * 16 + 8] = z1;
      __syncthreads();

      // ---- A fragment (16x32 f16 ISA layout) from LDS ----
      int r = wave * 16 + n0;                       // pixel row within tile
      const _Float16* ap = &Alds[r * 32 + kq * 8];  // K {0-7}|{8-15}
      v8h alo = *(const v8h*)ap;                    // + K {16-23}|{24-31}
      v8h ahi = *(const v8h*)(ap + 16);
      v16h afrag = SHUF16(alo, ahi);

      // ---- B fragments (32x16 f16 ISA layout) from packed weights ----
      const _Float16* wb =
          wp + ((size_t)tap * CN) * CN + half * 32 + kq * 8;
#pragma unroll
      for (int nb = 0; nb < 4; ++nb) {
        const _Float16* bp = wb + (size_t)(nb * 16 + n0) * CN;
        v8h blo = *(const v8h*)bp;
        v8h bhi = *(const v8h*)(bp + 16);
        v16h bfrag = SHUF16(blo, bhi);
        acc[nb] = __builtin_amdgcn_wmma_f32_16x16x32_f16(
            false, afrag, false, bfrag, (short)0, acc[nb], false, false);
      }
    }
  }

  // ---- epilogue: bias + ReLU, store NHWC f16 ----
  // C/D layout: VGPR i, lane l -> M = i + 8*(l>=16), N = l&15.
  int hq = lane >> 4;
  _Float16* dst = out + (((size_t)b * HN + h) * WN) * CN;
#pragma unroll
  for (int nb = 0; nb < 4; ++nb) {
    int co = nb * 16 + n0;
    float bv = bias[co];
#pragma unroll
    for (int i = 0; i < 8; ++i) {
      int m = wave * 16 + i + 8 * hq;
      float v = acc[nb][i] + bv;
      v = v > 0.f ? v : 0.f;
      dst[(size_t)(w0 + m) * CN + co] = (_Float16)v;
    }
  }
}

// ---------------------------------------------------------------------------
// 4a. rowpart[b][j][c] = sum_{w>=j} r[b,j,w,c]   (one block per (b,row))
// ---------------------------------------------------------------------------
__global__ __launch_bounds__(64) void rowpart_kernel(
    const _Float16* __restrict__ r, float* __restrict__ rowp) {
  int b = blockIdx.x >> 8;
  int h = blockIdx.x & 255;
  int c = threadIdx.x;
  const _Float16* base = r + (((size_t)b * HN + h) * WN) * CN + c;
  float s = 0.f;
  for (int w = h; w < WN; ++w) s += (float)base[(size_t)w * CN];
  rowp[((size_t)b * HN + h) * CN + c] = s;
}

// 4b. colpart[b][j][c] = sum_{h>j} r[b,h,j,c]   (block per (b, 4 cols))
__global__ __launch_bounds__(256) void colpart_kernel(
    const _Float16* __restrict__ r, float* __restrict__ colp) {
  int b  = blockIdx.x >> 6;
  int w0 = (blockIdx.x & 63) * 4;
  int c  = threadIdx.x & 63;
  int w  = w0 + (threadIdx.x >> 6);
  float s = 0.f;
  for (int h = w + 1; h < HN; ++h)
    s += (float)r[(((size_t)b * HN + h) * WN + w) * CN + c];
  colp[((size_t)b * HN + w) * CN + c] = s;
}

// 4c. Suffix-corner recurrence -> crop means[b][j][c], j = 0..128.
__global__ __launch_bounds__(64) void means_kernel(
    const float* __restrict__ rowp, const float* __restrict__ colp,
    float* __restrict__ means) {
  int b = blockIdx.x;
  int c = threadIdx.x;
  float S = 0.f;
  for (int j = HN - 1; j >= 0; --j) {
    size_t o = ((size_t)b * HN + j) * CN + c;
    S += rowp[o] + colp[o];
    if (j < NJ) {
      float side = (float)(HN - j);
      means[((size_t)b * NJ + j) * CN + c] = S / (side * side);
    }
  }
}

// ---------------------------------------------------------------------------
// 5. ECA gate: per (b,c), mean over crops of  m * sigmoid(conv1d_k3(m along c))
// ---------------------------------------------------------------------------
__global__ __launch_bounds__(64) void eca_gate_kernel(
    const float* __restrict__ means, const float* __restrict__ ew,
    float* __restrict__ gate) {
  int b = blockIdx.x;
  int c = threadIdx.x;
  float e0 = ew[0], e1 = ew[1], e2 = ew[2];
  float acc = 0.f;
  for (int j = 0; j < NJ; ++j) {
    const float* m = means + ((size_t)b * NJ + j) * CN;
    float left  = (c > 0)      ? m[c - 1] : 0.f;
    float mid   = m[c];
    float right = (c < CN - 1) ? m[c + 1] : 0.f;
    float y   = e0 * left + e1 * mid + e2 * right;
    float sig = 1.f / (1.f + __expf(-y));
    acc += mid * sig;
  }
  gate[b * CN + c] = acc / (float)NJ;
}

// ---------------------------------------------------------------------------
// 6. out = relu(r2 * gate + x), NCHW f32 (LDS-transpose r2; block per (b,h)).
// ---------------------------------------------------------------------------
__global__ __launch_bounds__(256) void final_kernel(
    const _Float16* __restrict__ r2, const float* __restrict__ x,
    const float* __restrict__ gate, float* __restrict__ out) {
  __shared__ _Float16 lds[WN * (CN + 2)];   // row stride 66 halves: no conflicts
  int b = blockIdx.x >> 8;
  int h = blockIdx.x & 255;
  int tid = threadIdx.x;
  const _Float16* src = r2 + (((size_t)b * HN + h) * WN) * CN;
#pragma unroll 4
  for (int it = 0; it < 64; ++it) {
    int idx = it * 256 + tid;                // coalesced NHWC read
    int w = idx >> 6;
    int c = idx & 63;
    lds[w * (CN + 2) + c] = src[idx];
  }
  __syncthreads();
  int w = tid;
  for (int c = 0; c < CN; ++c) {
    float g = gate[b * CN + c];
    size_t o = (((size_t)b * CN + c) * HN + h) * WN + w;  // coalesced NCHW
    float v = (float)lds[w * (CN + 2) + c] * g + x[o];
    out[o] = v > 0.f ? v : 0.f;
  }
}

// ---------------------------------------------------------------------------
// Launcher
// ---------------------------------------------------------------------------
extern "C" void kernel_launch(void* const* d_in, const int* in_sizes, int n_in,
                              void* d_out, int out_size, void* d_ws, size_t ws_size,
                              hipStream_t stream) {
  const float* x       = (const float*)d_in[0];
  const float* conv1_w = (const float*)d_in[1];
  const float* conv1_b = (const float*)d_in[2];
  const float* bn1_g   = (const float*)d_in[3];
  const float* bn1_b   = (const float*)d_in[4];
  const float* bn1_m   = (const float*)d_in[5];
  const float* bn1_v   = (const float*)d_in[6];
  const float* conv2_w = (const float*)d_in[7];
  const float* conv2_b = (const float*)d_in[8];
  const float* bn2_g   = (const float*)d_in[9];
  const float* bn2_b   = (const float*)d_in[10];
  const float* bn2_m   = (const float*)d_in[11];
  const float* bn2_v   = (const float*)d_in[12];
  const float* eca_w   = (const float*)d_in[13];
  float* out = (float*)d_out;

  char* ws = (char*)d_ws;
  _Float16* xh    = (_Float16*)(ws + OFF_XH);
  _Float16* r1h   = (_Float16*)(ws + OFF_R1);
  _Float16* r2h   = (_Float16*)(ws + OFF_R2);
  _Float16* wp1   = (_Float16*)(ws + OFF_WP1);
  _Float16* wp2   = (_Float16*)(ws + OFF_WP2);
  float*    bias1 = (float*)(ws + OFF_B1);
  float*    bias2 = (float*)(ws + OFF_B2);
  float*    rowp  = (float*)(ws + OFF_ROWP);
  float*    colp  = (float*)(ws + OFF_COLP);
  float*    means = (float*)(ws + OFF_MEANS);
  float*    gate  = (float*)(ws + OFF_GATE);

  // weight/bias folding (tiny)
  pack_w_kernel<<<(9 * CN * CN + 255) / 256, 256, 0, stream>>>(conv1_w, bn1_g, bn1_v, wp1);
  pack_w_kernel<<<(9 * CN * CN + 255) / 256, 256, 0, stream>>>(conv2_w, bn2_g, bn2_v, wp2);
  fold_bias_kernel<<<1, 64, 0, stream>>>(conv1_b, bn1_g, bn1_b, bn1_m, bn1_v, bias1);
  fold_bias_kernel<<<1, 64, 0, stream>>>(conv2_b, bn2_g, bn2_b, bn2_m, bn2_v, bias2);

  // layout conversion
  nchw_to_nhwc_kernel<<<BN * HN, 256, 0, stream>>>(x, xh);

  // conv1 -> r1, conv2 -> r2 (WMMA implicit GEMM, BN+ReLU fused)
  conv3x3_wmma_kernel<<<BN * HN * (WN / 128), 256, 0, stream>>>(xh,  wp1, bias1, r1h);
  conv3x3_wmma_kernel<<<BN * HN * (WN / 128), 256, 0, stream>>>(r1h, wp2, bias2, r2h);

  // cropped-pool means via suffix-corner decomposition
  rowpart_kernel<<<BN * HN, 64, 0, stream>>>(r2h, rowp);
  colpart_kernel<<<BN * (WN / 4), 256, 0, stream>>>(r2h, colp);
  means_kernel<<<BN, 64, 0, stream>>>(rowp, colp, means);

  // ECA gate + fused residual epilogue
  eca_gate_kernel<<<BN, 64, 0, stream>>>(means, eca_w, gate);
  final_kernel<<<BN * HN, 256, 0, stream>>>(r2h, x, gate, out);
}
